// MixtureOfBidders_59450937311879
// MI455X (gfx1250) — compile-verified, matrix operand order checked by reading, not run on hardware
//
#include <hip/hip_runtime.h>

// ---------------------------------------------------------------------------
// Problem constants (from the reference)
// ---------------------------------------------------------------------------
#define T_TOK 4096   // B*S tokens
#define HD    2048   // hidden
#define ID    7168   // intermediate
#define NE    8      // experts
#define NR    64     // LoRA rank
// SCALING = 0.25 folded into t (gate/up) and into down-LoRA epilogue.

typedef __attribute__((ext_vector_type(16))) __bf16 v16bf;
typedef __attribute__((ext_vector_type(8)))  __bf16 v8bf;
typedef __attribute__((ext_vector_type(8)))  float  v8f;

// ---------------------------------------------------------------------------
// WMMA helpers (CDNA5 wave32, v_wmma_f32_16x16x32_bf16)
// A-frag / B-frag: lane l holds row/col (l&15); K-halves selected by l>>4.
// C/D: VGPR v -> M = v + 8*(l>>4), N = l&15.
// ---------------------------------------------------------------------------
__device__ __forceinline__ v16bf frag_ld(const __bf16* p, int stride) {
  const int l = threadIdx.x & 31;
  const __bf16* q = p + (l & 15) * stride + ((l >> 4) << 3);
  v8bf lo = *(const v8bf*)q;
  v8bf hi = *(const v8bf*)(q + 16);
  return __builtin_shufflevector(lo, hi, 0,1,2,3,4,5,6,7,8,9,10,11,12,13,14,15);
}

__device__ __forceinline__ v8f wmma_bf16(v16bf a, v16bf b, v8f c) {
  return __builtin_amdgcn_wmma_f32_16x16x32_bf16(false, a, false, b, (short)0, c,
                                                 false, false);
}

// ---------------------------------------------------------------------------
// CDNA5 async global->LDS copy (16B per lane, tracked by ASYNCcnt).
// Low 32 bits of a generic __shared__ pointer are the LDS byte offset
// (ISA 10.2).
// ---------------------------------------------------------------------------
__device__ __forceinline__ void async_ld16(void* lds_dst, const void* gsrc) {
  unsigned off = (unsigned)(unsigned long long)lds_dst;
  asm volatile("global_load_async_to_lds_b128 %0, %1, off"
               :: "v"(off), "v"(gsrc) : "memory");
}
__device__ __forceinline__ void async_wait() {
  asm volatile("s_wait_asynccnt 0x0" ::: "memory");
}

// ---------------------------------------------------------------------------
// fp32 -> bf16 conversion (grid-stride)
// ---------------------------------------------------------------------------
__global__ void convert_kernel(const float* __restrict__ s, __bf16* __restrict__ d,
                               size_t n) {
  size_t i  = (size_t)blockIdx.x * blockDim.x + threadIdx.x;
  size_t st = (size_t)gridDim.x * blockDim.x;
  for (; i < n; i += st) d[i] = (__bf16)s[i];
}

// ---------------------------------------------------------------------------
// Fused transpose + fp32->bf16:  in [R][C] f32 (batched)  ->  out [C][R] bf16
// All B operands of the WMMA GEMMs are pre-transposed to n-major so tile
// staging is pure contiguous 16B copies (async global->LDS).
// ---------------------------------------------------------------------------
__global__ __launch_bounds__(256)
void transpose_convert_kernel(const float* __restrict__ in, __bf16* __restrict__ out,
                              int R, int C) {
  __shared__ float t[32][33];
  const int c0 = blockIdx.x * 32, r0 = blockIdx.y * 32;
  const size_t b = (size_t)blockIdx.z * R * C;
  const int tx = threadIdx.x, ty = threadIdx.y;   // (32, 8)
#pragma unroll
  for (int j = 0; j < 4; ++j) {
    int r = ty + j * 8;
    t[r][tx] = in[b + (size_t)(r0 + r) * C + c0 + tx];
  }
  __syncthreads();
#pragma unroll
  for (int j = 0; j < 4; ++j) {
    int c = ty + j * 8;
    out[b + (size_t)(c0 + c) * R + r0 + tx] = (__bf16)t[tx][c];
  }
}

// ---------------------------------------------------------------------------
// Routing: per-token confidences for 8 experts, top-2, softmax over top bids.
// ---------------------------------------------------------------------------
__global__ __launch_bounds__(256)
void routing_kernel(const float* __restrict__ x, const float* __restrict__ wealth,
                    const float* __restrict__ cW, const float* __restrict__ cb,
                    int* __restrict__ sel, float* __restrict__ wgt) {
  const int t = blockIdx.x;
  const float* xr = x + (size_t)t * HD;
  __shared__ float red[256][NE];
  float part[NE];
#pragma unroll
  for (int e = 0; e < NE; ++e) part[e] = 0.f;
  for (int h = threadIdx.x; h < HD; h += 256) {
    float xv = xr[h];
#pragma unroll
    for (int e = 0; e < NE; ++e) part[e] += xv * cW[e * HD + h];
  }
#pragma unroll
  for (int e = 0; e < NE; ++e) red[threadIdx.x][e] = part[e];
  __syncthreads();
  for (int s2 = 128; s2 > 0; s2 >>= 1) {
    if (threadIdx.x < (unsigned)s2)
#pragma unroll
      for (int e = 0; e < NE; ++e) red[threadIdx.x][e] += red[threadIdx.x + s2][e];
    __syncthreads();
  }
  if (threadIdx.x == 0) {
    float bids[NE];
#pragma unroll
    for (int e = 0; e < NE; ++e) {
      float c = 1.f / (1.f + __expf(-(red[0][e] + cb[e])));
      bids[e] = c * wealth[e];
    }
    int i0 = 0;
#pragma unroll
    for (int e = 1; e < NE; ++e) if (bids[e] > bids[i0]) i0 = e;
    int i1 = -1;
#pragma unroll
    for (int e = 0; e < NE; ++e)
      if (e != i0 && (i1 < 0 || bids[e] > bids[i1])) i1 = e;
    float ex = __expf(bids[i1] - bids[i0]);   // <= 1
    float w0 = 1.f / (1.f + ex);
    sel[t * 2 + 0] = i0;  sel[t * 2 + 1] = i1;
    wgt[t * 2 + 0] = w0;  wgt[t * 2 + 1] = ex * w0;
  }
}

// ---------------------------------------------------------------------------
// Expert grouping: counts -> offsets -> scatter (8192 assignments)
// ---------------------------------------------------------------------------
__global__ void count_kernel(const int* __restrict__ sel, int* __restrict__ counts) {
  int i = blockIdx.x * blockDim.x + threadIdx.x;
  if (i < T_TOK * 2) atomicAdd(&counts[sel[i]], 1);
}
__global__ void offsets_kernel(const int* __restrict__ counts,
                               int* __restrict__ offsets, int* __restrict__ cursors) {
  if (threadIdx.x == 0 && blockIdx.x == 0) {
    int o = 0;
    for (int e = 0; e < NE; ++e) { offsets[e] = o; cursors[e] = o; o += counts[e]; }
  }
}
__global__ void scatter_kernel(const int* __restrict__ sel, const float* __restrict__ wgt,
                               int* __restrict__ cursors, int* __restrict__ ptoken,
                               float* __restrict__ pwgt) {
  int i = blockIdx.x * blockDim.x + threadIdx.x;
  if (i < T_TOK * 2) {
    int e = sel[i];
    int row = atomicAdd(&cursors[e], 1);
    ptoken[row] = i >> 1;
    pwgt[row]   = wgt[i];
  }
}

// ---------------------------------------------------------------------------
// Generic WMMA GEMM: C[128x64 tile] = A * Bt^T  (* scale)
// A row-major (bf16, or f32 converted during LDS staging); Bt is the
// PRE-TRANSPOSED operand, n-major [N][K] with row stride ldb.
// ---------------------------------------------------------------------------
template <bool A_F32, bool OUT_BF16>
__global__ __launch_bounds__(256)
void gemm_kernel(const void* __restrict__ Ap, int lda,
                 const __bf16* __restrict__ Bt, int ldb,
                 void* __restrict__ Cp, int ldc, int Kdim, float scale) {
  __shared__ alignas(16) __bf16 As[128 * 40];
  __shared__ alignas(16) __bf16 Bs[64 * 40];
  const int tid = threadIdx.x;
  const int m0 = blockIdx.y * 128;
  const int n0 = blockIdx.x * 64;
  const int l = tid & 31, wv = tid >> 5;
  const int wm = wv >> 1, wn = wv & 1;
  v8f acc[2][2] = {};
  for (int k0 = 0; k0 < Kdim; k0 += 32) {
    __syncthreads();
#pragma unroll
    for (int j = 0; j < 2; ++j) {           // A tile: 128x32, 512 pieces of 8
      int p = tid + j * 256;
      int row = p >> 2, cp = (p & 3) << 3;
      if constexpr (A_F32) {
        const float* src = (const float*)Ap + (size_t)(m0 + row) * lda + k0 + cp;
        __builtin_prefetch(src + 64, 0, 1);
        float4 f0 = *(const float4*)src;
        float4 f1 = *(const float4*)(src + 4);
        v8bf o;
        o[0] = (__bf16)f0.x; o[1] = (__bf16)f0.y; o[2] = (__bf16)f0.z; o[3] = (__bf16)f0.w;
        o[4] = (__bf16)f1.x; o[5] = (__bf16)f1.y; o[6] = (__bf16)f1.z; o[7] = (__bf16)f1.w;
        *(v8bf*)&As[row * 40 + cp] = o;
      } else {
        async_ld16(&As[row * 40 + cp],
                   (const __bf16*)Ap + (size_t)(m0 + row) * lda + k0 + cp);
      }
    }
    {                                        // B tile: 64 n-rows x 32 k
      int nr = tid >> 2, kc = (tid & 3) << 3;
      async_ld16(&Bs[nr * 40 + kc], Bt + (size_t)(n0 + nr) * ldb + k0 + kc);
    }
    async_wait();
    __syncthreads();
    v16bf af[2], bfv[2];
    af[0] = frag_ld(&As[(wm * 32) * 40], 40);
    af[1] = frag_ld(&As[(wm * 32 + 16) * 40], 40);
    bfv[0] = frag_ld(&Bs[(wn * 32) * 40], 40);
    bfv[1] = frag_ld(&Bs[(wn * 32 + 16) * 40], 40);
#pragma unroll
    for (int tm = 0; tm < 2; ++tm)
#pragma unroll
      for (int tn = 0; tn < 2; ++tn)
        acc[tm][tn] = wmma_bf16(af[tm], bfv[tn], acc[tm][tn]);
  }
#pragma unroll
  for (int tm = 0; tm < 2; ++tm)
#pragma unroll
    for (int tn = 0; tn < 2; ++tn)
#pragma unroll
      for (int v = 0; v < 8; ++v) {
        int m = m0 + wm * 32 + tm * 16 + v + ((l >> 4) << 3);
        int n = n0 + wn * 32 + tn * 16 + (l & 15);
        float val = acc[tm][tn][v] * scale;
        if constexpr (OUT_BF16) ((__bf16*)Cp)[(size_t)m * ldc + n] = (__bf16)val;
        else                    ((float*)Cp)[(size_t)m * ldc + n] = val;
      }
}

// ---------------------------------------------------------------------------
// Grouped fused gate/up kernel (per expert, gathered token rows).
//  g = x@Wg + t_g@gate_B   (t_g pre-scaled by 0.25)  -- K = 2048 + 64
//  u = x@Wu + t_u@up_B
//  wh = w * silu(g) * u  -> atomicAdd into hbar[token]
//  r[assignment] += wh_tile @ down_A[e] slice        (WMMA, atomic partials)
// All B operands are pre-transposed (n-major).
// ---------------------------------------------------------------------------
__global__ __launch_bounds__(256)
void moe_gateup_kernel(const __bf16* __restrict__ xbf,
                       const __bf16* __restrict__ WgT,  // [ID][HD]
                       const __bf16* __restrict__ WuT,  // [ID][HD]
                       const __bf16* __restrict__ gBT,  // [E][ID][NR]
                       const __bf16* __restrict__ uBT,  // [E][ID][NR]
                       const __bf16* __restrict__ tg, const __bf16* __restrict__ tu,
                       const __bf16* __restrict__ dAT,  // [E][NR][ID]
                       const int* __restrict__ ptoken, const float* __restrict__ pwgt,
                       const int* __restrict__ offsets, const int* __restrict__ counts,
                       float* __restrict__ hbar, float* __restrict__ rbuf) {
  const int e = blockIdx.z;
  const int cnt = counts[e], base = offsets[e];
  const int tile0 = blockIdx.y * 128;
  if (tile0 >= cnt) return;
  const int n0 = blockIdx.x * 64;

  __shared__ alignas(16) __bf16 As [128 * 40];
  __shared__ alignas(16) __bf16 As2[128 * 40];
  __shared__ alignas(16) __bf16 Bg [64 * 40];
  __shared__ alignas(16) __bf16 Bu [64 * 40];
  __shared__ alignas(16) __bf16 Ht [128 * 72];
  __shared__ alignas(16) __bf16 Br [64 * 72];
  __shared__ int   s_tok[128];
  __shared__ float s_w[128];

  const int tid = threadIdx.x;
  if (tid < 128) {
    int row = tile0 + tid;
    bool valid = row < cnt;
    s_tok[tid] = valid ? ptoken[base + row] : 0;
    s_w[tid]   = valid ? pwgt[base + row]   : 0.f;
  }
  __syncthreads();

  const int l = tid & 31, wv = tid >> 5, wm = wv >> 1, wn = wv & 1;
  v8f ag[2][2] = {}, au[2][2] = {};

  for (int kk = 0; kk < (HD + NR) / 32; ++kk) {   // 64 main + 2 LoRA steps
    const int k0 = kk * 32;
    const bool lora = (k0 >= HD);
    const int kq = lora ? (k0 - HD) : k0;
    __syncthreads();
#pragma unroll
    for (int j = 0; j < 2; ++j) {                 // A tiles (gathered rows)
      int p = tid + j * 256;
      int row = p >> 2, cp = (p & 3) << 3;
      int tok = s_tok[row];
      if (!lora) {
        async_ld16(&As[row * 40 + cp], xbf + (size_t)tok * HD + k0 + cp);
      } else {
        async_ld16(&As [row * 40 + cp], tg + ((size_t)tok * NE + e) * NR + kq + cp);
        async_ld16(&As2[row * 40 + cp], tu + ((size_t)tok * NE + e) * NR + kq + cp);
      }
    }
    {                                             // B tiles (n-major, contiguous)
      int nr = tid >> 2, kc = (tid & 3) << 3;
      const __bf16 *bg, *bu;
      if (!lora) {
        bg = WgT + (size_t)(n0 + nr) * HD + k0 + kc;
        bu = WuT + (size_t)(n0 + nr) * HD + k0 + kc;
      } else {
        bg = gBT + ((size_t)e * ID + n0 + nr) * NR + kq + kc;
        bu = uBT + ((size_t)e * ID + n0 + nr) * NR + kq + kc;
      }
      async_ld16(&Bg[nr * 40 + kc], bg);
      async_ld16(&Bu[nr * 40 + kc], bu);
    }
    async_wait();
    __syncthreads();
    v16bf a1[2], a2[2], bgf[2], buf2[2];
    a1[0] = frag_ld(&As[(wm * 32) * 40], 40);
    a1[1] = frag_ld(&As[(wm * 32 + 16) * 40], 40);
    if (lora) {
      a2[0] = frag_ld(&As2[(wm * 32) * 40], 40);
      a2[1] = frag_ld(&As2[(wm * 32 + 16) * 40], 40);
    } else {
      a2[0] = a1[0]; a2[1] = a1[1];
    }
    bgf[0] = frag_ld(&Bg[(wn * 32) * 40], 40);
    bgf[1] = frag_ld(&Bg[(wn * 32 + 16) * 40], 40);
    buf2[0] = frag_ld(&Bu[(wn * 32) * 40], 40);
    buf2[1] = frag_ld(&Bu[(wn * 32 + 16) * 40], 40);
#pragma unroll
    for (int tm = 0; tm < 2; ++tm)
#pragma unroll
      for (int tn = 0; tn < 2; ++tn) {
        ag[tm][tn] = wmma_bf16(a1[tm], bgf[tn], ag[tm][tn]);
        au[tm][tn] = wmma_bf16(a2[tm], buf2[tn], au[tm][tn]);
      }
  }

  // epilogue: silu(g)*u, weight, scatter into hbar, stash wh tile in LDS
#pragma unroll
  for (int tm = 0; tm < 2; ++tm)
#pragma unroll
    for (int tn = 0; tn < 2; ++tn)
#pragma unroll
      for (int v = 0; v < 8; ++v) {
        int mi = wm * 32 + tm * 16 + v + ((l >> 4) << 3);
        int ni = wn * 32 + tn * 16 + (l & 15);
        float g = ag[tm][tn][v], u = au[tm][tn][v];
        float h = g / (1.f + __expf(-g)) * u;
        float wh = s_w[mi] * h;
        Ht[mi * 72 + ni] = (__bf16)wh;
        if (tile0 + mi < cnt)
          atomicAdd(&hbar[(size_t)s_tok[mi] * ID + n0 + ni], wh);
      }

  // down-LoRA projection: r_tile = wh_tile @ down_A[e][n0:n0+64, :]
  // B operand from dAT [E][NR][ID]: row = R-index, cols = I-slice (contiguous)
#pragma unroll
  for (int j = 0; j < 2; ++j) {
    int p = tid + j * 256;                        // 64x64 = 512 pieces
    int nr = p >> 3, kc = (p & 7) << 3;
    async_ld16(&Br[nr * 72 + kc], dAT + ((size_t)e * NR + nr) * ID + n0 + kc);
  }
  async_wait();
  __syncthreads();
  v8f ar[2][2] = {};
#pragma unroll
  for (int ks = 0; ks < 2; ++ks) {
    v16bf ha[2], bb[2];
    ha[0] = frag_ld(&Ht[(wm * 32) * 72 + ks * 32], 72);
    ha[1] = frag_ld(&Ht[(wm * 32 + 16) * 72 + ks * 32], 72);
    bb[0] = frag_ld(&Br[(wn * 32) * 72 + ks * 32], 72);
    bb[1] = frag_ld(&Br[(wn * 32 + 16) * 72 + ks * 32], 72);
#pragma unroll
    for (int tm = 0; tm < 2; ++tm)
#pragma unroll
      for (int tn = 0; tn < 2; ++tn)
        ar[tm][tn] = wmma_bf16(ha[tm], bb[tn], ar[tm][tn]);
  }
#pragma unroll
  for (int tm = 0; tm < 2; ++tm)
#pragma unroll
    for (int tn = 0; tn < 2; ++tn)
#pragma unroll
      for (int v = 0; v < 8; ++v) {
        int mi = wm * 32 + tm * 16 + v + ((l >> 4) << 3);
        int rc = wn * 32 + tn * 16 + (l & 15);
        if (tile0 + mi < cnt)
          atomicAdd(&rbuf[(size_t)(base + tile0 + mi) * NR + rc], ar[tm][tn][v]);
      }
}

// ---------------------------------------------------------------------------
// Grouped down-LoRA output: out[token] += 0.25 * r[assignment] @ down_B[e]
// B operand from dBT [E][HD][NR] (n-major).
// ---------------------------------------------------------------------------
__global__ __launch_bounds__(256)
void moe_dlora_kernel(const float* __restrict__ rbuf, const __bf16* __restrict__ dBT,
                      const int* __restrict__ ptoken, const int* __restrict__ offsets,
                      const int* __restrict__ counts, float* __restrict__ out) {
  const int e = blockIdx.z;
  const int cnt = counts[e], base = offsets[e];
  const int tile0 = blockIdx.y * 128;
  if (tile0 >= cnt) return;
  const int n0 = blockIdx.x * 64;
  __shared__ alignas(16) __bf16 As[128 * 72];
  __shared__ alignas(16) __bf16 Bs[64 * 72];
  __shared__ int s_tok[128];
  const int tid = threadIdx.x;
  if (tid < 128) {
    int row = tile0 + tid;
    s_tok[tid] = (row < cnt) ? ptoken[base + row] : 0;
  }
#pragma unroll
  for (int j = 0; j < 4; ++j) {                   // A: 128x64 f32 -> bf16
    int p = tid + j * 256;
    int row = p >> 3, cp = (p & 7) << 3;
    int rr = tile0 + row; if (rr >= cnt) rr = cnt - 1;
    const float* s = rbuf + (size_t)(base + rr) * NR + cp;
    float4 f0 = *(const float4*)s;
    float4 f1 = *(const float4*)(s + 4);
    v8bf o;
    o[0] = (__bf16)f0.x; o[1] = (__bf16)f0.y; o[2] = (__bf16)f0.z; o[3] = (__bf16)f0.w;
    o[4] = (__bf16)f1.x; o[5] = (__bf16)f1.y; o[6] = (__bf16)f1.z; o[7] = (__bf16)f1.w;
    *(v8bf*)&As[row * 72 + cp] = o;
  }
#pragma unroll
  for (int j = 0; j < 2; ++j) {                   // B: 64 n-rows x 64 K (=NR)
    int p = tid + j * 256;
    int nr = p >> 3, kc = (p & 7) << 3;
    async_ld16(&Bs[nr * 72 + kc], dBT + ((size_t)e * HD + n0 + nr) * NR + kc);
  }
  async_wait();
  __syncthreads();
  const int l = tid & 31, wv = tid >> 5, wm = wv >> 1, wn = wv & 1;
  v8f acc[2][2] = {};
#pragma unroll
  for (int ks = 0; ks < 2; ++ks) {
    v16bf af[2], bf2[2];
    af[0] = frag_ld(&As[(wm * 32) * 72 + ks * 32], 72);
    af[1] = frag_ld(&As[(wm * 32 + 16) * 72 + ks * 32], 72);
    bf2[0] = frag_ld(&Bs[(wn * 32) * 72 + ks * 32], 72);
    bf2[1] = frag_ld(&Bs[(wn * 32 + 16) * 72 + ks * 32], 72);
#pragma unroll
    for (int tm = 0; tm < 2; ++tm)
#pragma unroll
      for (int tn = 0; tn < 2; ++tn)
        acc[tm][tn] = wmma_bf16(af[tm], bf2[tn], acc[tm][tn]);
  }
#pragma unroll
  for (int tm = 0; tm < 2; ++tm)
#pragma unroll
    for (int tn = 0; tn < 2; ++tn)
#pragma unroll
      for (int v = 0; v < 8; ++v) {
        int mi = wm * 32 + tm * 16 + v + ((l >> 4) << 3);
        int ni = wn * 32 + tn * 16 + (l & 15);
        if (tile0 + mi < cnt)
          atomicAdd(&out[(size_t)s_tok[mi] * HD + n0 + ni], 0.25f * acc[tm][tn][v]);
      }
}

// ---------------------------------------------------------------------------
// Host launcher
// ---------------------------------------------------------------------------
extern "C" void kernel_launch(void* const* d_in, const int* in_sizes, int n_in,
                              void* d_out, int out_size, void* d_ws, size_t ws_size,
                              hipStream_t stream) {
  (void)in_sizes; (void)n_in; (void)out_size; (void)ws_size;
  const float* x      = (const float*)d_in[0];
  const float* wealth = (const float*)d_in[1];
  const float* confW  = (const float*)d_in[2];
  const float* confb  = (const float*)d_in[3];
  const float* baseG  = (const float*)d_in[4];
  const float* baseU  = (const float*)d_in[5];
  const float* baseD  = (const float*)d_in[6];
  const float* gateA  = (const float*)d_in[7];
  const float* gateB  = (const float*)d_in[8];
  const float* upA    = (const float*)d_in[9];
  const float* upB    = (const float*)d_in[10];
  const float* downA  = (const float*)d_in[11];
  const float* downB  = (const float*)d_in[12];
  float* out = (float*)d_out;

  char* w = (char*)d_ws;
  size_t off = 0;
  auto alloc = [&](size_t bytes) -> void* {
    void* p = w + off;
    off = (off + bytes + 255) & ~(size_t)255;
    return p;
  };
  __bf16* xbf  = (__bf16*)alloc((size_t)T_TOK * HD * 2);
  __bf16* WgT  = (__bf16*)alloc((size_t)HD * ID * 2);
  __bf16* WuT  = (__bf16*)alloc((size_t)HD * ID * 2);
  __bf16* WdT  = (__bf16*)alloc((size_t)ID * HD * 2);
  __bf16* gAT  = (__bf16*)alloc((size_t)NE * HD * NR * 2);
  __bf16* uAT  = (__bf16*)alloc((size_t)NE * HD * NR * 2);
  __bf16* gBT  = (__bf16*)alloc((size_t)NE * NR * ID * 2);
  __bf16* uBT  = (__bf16*)alloc((size_t)NE * NR * ID * 2);
  __bf16* dAT  = (__bf16*)alloc((size_t)NE * ID * NR * 2);
  __bf16* dBT  = (__bf16*)alloc((size_t)NE * NR * HD * 2);
  __bf16* tg   = (__bf16*)alloc((size_t)T_TOK * NE * NR * 2);
  __bf16* tu   = (__bf16*)alloc((size_t)T_TOK * NE * NR * 2);
  float*  hbar = (float*)alloc((size_t)T_TOK * ID * 4);
  float*  rbuf = (float*)alloc((size_t)T_TOK * 2 * NR * 4);
  int*    sel  = (int*)alloc((size_t)T_TOK * 2 * 4);
  float*  wgt  = (float*)alloc((size_t)T_TOK * 2 * 4);
  int*    ptok = (int*)alloc((size_t)T_TOK * 2 * 4);
  float*  pwgt = (float*)alloc((size_t)T_TOK * 2 * 4);
  int*    counts  = (int*)alloc(NE * 4);
  int*    offsets = (int*)alloc(NE * 4);
  int*    cursors = (int*)alloc(NE * 4);

  hipMemsetAsync(counts, 0, NE * 4, stream);
  hipMemsetAsync(hbar, 0, (size_t)T_TOK * ID * 4, stream);
  hipMemsetAsync(rbuf, 0, (size_t)T_TOK * 2 * NR * 4, stream);

  // activations: plain convert
  convert_kernel<<<1024, 256, 0, stream>>>(x, xbf, (size_t)T_TOK * HD);

  // weights: fused transpose + convert to n-major bf16
  dim3 tb(32, 8);
  transpose_convert_kernel<<<dim3(ID/32, HD/32, 1),  tb, 0, stream>>>(baseG, WgT, HD, ID);
  transpose_convert_kernel<<<dim3(ID/32, HD/32, 1),  tb, 0, stream>>>(baseU, WuT, HD, ID);
  transpose_convert_kernel<<<dim3(HD/32, ID/32, 1),  tb, 0, stream>>>(baseD, WdT, ID, HD);
  transpose_convert_kernel<<<dim3(NR/32, HD/32, NE), tb, 0, stream>>>(gateA, gAT, HD, NR);
  transpose_convert_kernel<<<dim3(NR/32, HD/32, NE), tb, 0, stream>>>(upA,   uAT, HD, NR);
  transpose_convert_kernel<<<dim3(ID/32, NR/32, NE), tb, 0, stream>>>(gateB, gBT, NR, ID);
  transpose_convert_kernel<<<dim3(ID/32, NR/32, NE), tb, 0, stream>>>(upB,   uBT, NR, ID);
  transpose_convert_kernel<<<dim3(NR/32, ID/32, NE), tb, 0, stream>>>(downA, dAT, ID, NR);
  transpose_convert_kernel<<<dim3(HD/32, NR/32, NE), tb, 0, stream>>>(downB, dBT, NR, HD);

  // routing + expert grouping
  routing_kernel<<<T_TOK, 256, 0, stream>>>(x, wealth, confW, confb, sel, wgt);
  count_kernel<<<32, 256, 0, stream>>>(sel, counts);
  offsets_kernel<<<1, 1, 0, stream>>>(counts, offsets, cursors);
  scatter_kernel<<<32, 256, 0, stream>>>(sel, wgt, cursors, ptok, pwgt);

  // LoRA A-projections (dense, tiny): t = 0.25 * x @ A_e, bf16 out
  for (int e = 0; e < NE; ++e) {
    gemm_kernel<false, true><<<dim3(1, T_TOK / 128), 256, 0, stream>>>(
        xbf, HD, gAT + (size_t)e * NR * HD, HD,
        (void*)(tg + e * NR), NE * NR, HD, 0.25f);
    gemm_kernel<false, true><<<dim3(1, T_TOK / 128), 256, 0, stream>>>(
        xbf, HD, uAT + (size_t)e * NR * HD, HD,
        (void*)(tu + e * NR), NE * NR, HD, 0.25f);
  }

  // fused grouped gate/up + silu + hbar + down-LoRA projection
  moe_gateup_kernel<<<dim3(ID / 64, T_TOK / 128, NE), 256, 0, stream>>>(
      xbf, WgT, WuT, gBT, uBT, tg, tu, dAT, ptok, pwgt, offsets, counts,
      hbar, rbuf);

  // shared down projection: out = hbar @ Wd  (WdT is n-major [HD][ID])
  gemm_kernel<true, false><<<dim3(HD / 64, T_TOK / 128), 256, 0, stream>>>(
      hbar, ID, WdT, ID, (void*)out, HD, ID, 1.0f);

  // grouped down-LoRA output: out += 0.25 * r @ down_B[e]
  moe_dlora_kernel<<<dim3(HD / 64, T_TOK / 128, NE), 256, 0, stream>>>(
      rbuf, dBT, ptok, offsets, counts, out);
}
